// SelfAttention_50543175139866
// MI455X (gfx1250) — matrix operand order repaired and checked by reading, not compiled, not run on hardware
//
#include <hip/hip_runtime.h>
#include <hip/hip_bf16.h>

typedef __attribute__((ext_vector_type(16))) _Float16 v16h;
typedef __attribute__((ext_vector_type(8)))  _Float16 v8h;
typedef __attribute__((ext_vector_type(8)))  float    v8f;
typedef __attribute__((ext_vector_type(4)))  unsigned int v4u;
typedef __attribute__((ext_vector_type(8)))  int v8i;
typedef __attribute__((ext_vector_type(4)))  int v4i;

constexpr int Bn   = 8;
constexpr int Cc   = 256;
constexpr int Ll   = 2048;
constexpr int DK   = 256;
constexpr int KDIM = Cc * 3;   // 768, im2col K dimension

// ---- fragment helpers (CDNA5 16-bit WMMA layouts, wave32) ----------------

// A-matrix 16x32 f16: lane m<16 -> row m, K halves {0..7} then {16..23};
// lane m+16 -> same row, K halves {8..15} then {24..31}.
__device__ inline v16h load_frag_a(const _Float16* row, int lane) {
  const int base = (lane < 16) ? 0 : 8;
  v8h lo = *(const v8h*)(row + base);
  v8h hi = *(const v8h*)(row + base + 16);
  v16h a;
#pragma unroll
  for (int i = 0; i < 8; i++) { a[i] = lo[i]; a[i + 8] = hi[i]; }
  return a;
}

// B-matrix 32x16 f16: lane n holds column n%16; K halves {0..7},{16..23}
// (lanes 0-15) or {8..15},{24..31} (lanes 16-31).  `col` must already point at
// [column][k0] in a K-contiguous layout.
__device__ inline v16h load_frag_b(const _Float16* col, int lane) {
  const _Float16* p = col + ((lane < 16) ? 0 : 8);
  v8h lo = *(const v8h*)(p);
  v8h hi = *(const v8h*)(p + 16);
  v16h b;
#pragma unroll
  for (int i = 0; i < 8; i++) { b[i] = lo[i]; b[i + 8] = hi[i]; }
  return b;
}

__device__ inline v8f wmma16(v16h a, v16h b, v8f c) {
  return __builtin_amdgcn_wmma_f32_16x16x32_f16(
      /*neg_a=*/false, a, /*neg_b=*/false, b,
      /*c_mod=*/(short)0, c, /*reuse_a=*/false, /*reuse_b=*/false);
}

// ---- Tensor Data Mover: 2D tile (global -> LDS), data_size = 2 bytes -----
// D# packing per CDNA5 ISA; groups 2/3 zero.
// This toolchain's builtin takes 6 args (g0, g1, g2, g3, g4, cpol).
__device__ inline void tdm_load_2d(unsigned int lds_addr, unsigned long long gaddr,
                                   unsigned int tensor_d0, unsigned int tensor_d1,
                                   unsigned int tile_d0, unsigned int tile_d1,
                                   unsigned int stride0) {
  v4u g0;
  g0[0] = 1u;                                   // count=1, user descriptor
  g0[1] = lds_addr;                             // LDS byte address
  g0[2] = (unsigned int)gaddr;                  // global_addr[31:0]
  g0[3] = (unsigned int)((gaddr >> 32) & 0x1FFFFFFu) | (2u << 30);  // [56:32] | type=2
  v8i g1;
  g1[0] = (int)0x00010000u;                     // data_size=1 (2 bytes)
  g1[1] = (int)((tensor_d0 & 0xFFFFu) << 16);   // tensor_dim0[15:0]
  g1[2] = (int)((tensor_d0 >> 16) | ((tensor_d1 & 0xFFFFu) << 16));
  g1[3] = (int)((tensor_d1 >> 16) | (tile_d0 << 16));   // tile_dim0
  g1[4] = (int)(tile_d1 & 0xFFFFu);             // tile_dim1 (tile_dim2 = 0)
  g1[5] = (int)stride0;                         // tensor_dim0_stride[31:0]
  g1[6] = 0;                                    // stride0[47:32]=0, stride1[15:0]=0
  g1[7] = 0;
  v4i gz4 = {0, 0, 0, 0};
  v8i gz8 = {0, 0, 0, 0, 0, 0, 0, 0};
  __builtin_amdgcn_tensor_load_to_lds(g0, g1, gz4, gz4, gz8, 0);
}

// ---- pack kernels --------------------------------------------------------

// im2col: xim[b][l][k], k = c*3+dl, value = x[b][c][l+dl-1] (zero padded)
__global__ __launch_bounds__(256) void pack_x_kernel(
    const float* __restrict__ x, _Float16* __restrict__ xim) {
  const size_t total = (size_t)Bn * Ll * KDIM;
  for (size_t i = (size_t)blockIdx.x * blockDim.x + threadIdx.x; i < total;
       i += (size_t)gridDim.x * blockDim.x) {
    int k = (int)(i % KDIM);
    size_t rest = i / KDIM;
    int l = (int)(rest % Ll);
    int b = (int)(rest / Ll);
    int c = k / 3, dl = k % 3;
    int ls = l + dl - 1;
    float v = (ls >= 0 && ls < Ll)
                  ? x[((size_t)b * Cc + c) * Ll + ls]
                  : 0.0f;
    xim[i] = (_Float16)v;
  }
}

// Wh[mat][d][k] f16 <- {Wq,Wk,Wv} f32 (flat copy: (C,KS) flattens to k=c*3+dl)
__global__ __launch_bounds__(256) void pack_w_kernel(
    const float* __restrict__ Wq, const float* __restrict__ Wk,
    const float* __restrict__ Wv, _Float16* __restrict__ Wh) {
  const size_t n = (size_t)DK * KDIM;
  const size_t total = 3 * n;
  for (size_t i = (size_t)blockIdx.x * blockDim.x + threadIdx.x; i < total;
       i += (size_t)gridDim.x * blockDim.x) {
    const float* src = (i < n) ? Wq : (i < 2 * n) ? Wk : Wv;
    Wh[i] = (_Float16)src[i % n];
  }
}

// ---- conv-as-GEMM: (256 x 768) x (768 x 2048) per (batch, matrix) --------
// grid = (L/64, DK/64, 3*B), block = 256 (8 waves). Wave w: m-subtile w%4,
// n-subtiles 2*(w/4)+{0,1}. Q,K -> [b][l][d]; V -> [b][d][l].
__global__ __launch_bounds__(256) void conv_gemm_kernel(
    const _Float16* __restrict__ Wh, const _Float16* __restrict__ xim,
    const float* __restrict__ bq, const float* __restrict__ bk,
    const float* __restrict__ bv,
    _Float16* __restrict__ Qh, _Float16* __restrict__ Kh,
    _Float16* __restrict__ Vh) {
  const int lane = threadIdx.x & 31;
  const int wave = threadIdx.x >> 5;
  const int mw = wave & 3;
  const int nw = wave >> 2;
  const int b   = blockIdx.z / 3;
  const int mat = blockIdx.z % 3;
  const int d0 = blockIdx.y * 64 + mw * 16;
  const int l0 = blockIdx.x * 64 + nw * 32;

  const _Float16* arow  = Wh + ((size_t)mat * DK + d0 + (lane & 15)) * KDIM;
  const _Float16* xcol0 = xim + ((size_t)b * Ll + l0 + (lane & 15)) * KDIM;
  const _Float16* xcol1 = xcol0 + (size_t)16 * KDIM;

  v8f acc0 = {}; v8f acc1 = {};
#pragma unroll 4
  for (int kk = 0; kk < KDIM; kk += 32) {
    v16h a  = load_frag_a(arow + kk, lane);
    v16h b0 = load_frag_b(xcol0 + kk, lane);
    v16h b1 = load_frag_b(xcol1 + kk, lane);
    acc0 = wmma16(a, b0, acc0);
    acc1 = wmma16(a, b1, acc1);
  }

  const float* bias = (mat == 0) ? bq : (mat == 1) ? bk : bv;
  const int hi = (lane >= 16) ? 8 : 0;
  float bb[8];
#pragma unroll
  for (int r = 0; r < 8; r++) bb[r] = bias[d0 + hi + r];

  if (mat < 2) {
    _Float16* O = (mat == 0) ? Qh : Kh;  // [b][l][d]
#pragma unroll
    for (int nt = 0; nt < 2; nt++) {
      v8f acc = nt ? acc1 : acc0;
      int l = l0 + nt * 16 + (lane & 15);
      v8h h;
#pragma unroll
      for (int r = 0; r < 8; r++) h[r] = (_Float16)(acc[r] + bb[r]);
      *(v8h*)(O + ((size_t)b * Ll + l) * DK + d0 + hi) = h;
    }
  } else {  // V -> [b][d][l]
#pragma unroll
    for (int nt = 0; nt < 2; nt++) {
      v8f acc = nt ? acc1 : acc0;
      int l = l0 + nt * 16 + (lane & 15);
#pragma unroll
      for (int r = 0; r < 8; r++)
        Vh[((size_t)b * DK + d0 + hi + r) * Ll + l] = (_Float16)(acc[r] + bb[r]);
    }
  }
}

// ---- flash attention -----------------------------------------------------
// grid = B * (L/64) blocks of 128 threads (4 waves, wave -> its own q-tile).
// K/V tiles are staged into LDS by the Tensor Data Mover (double-buffered,
// wave 0 issues + waits TENSORcnt, block barrier publishes to all waves).
__global__ __launch_bounds__(128) void attn_kernel(
    const _Float16* __restrict__ Qh, const _Float16* __restrict__ Kh,
    const _Float16* __restrict__ Vh, float* __restrict__ H) {
  __shared__ __align__(16) _Float16 Kt[2][32 * 256];   // [key][d]
  __shared__ __align__(16) _Float16 Vt[2][256 * 32];   // [d][key]
  __shared__ __align__(16) _Float16 pLds[4][16 * 32];
  const int lane = threadIdx.x & 31;
  const int wave = threadIdx.x >> 5;
  const int b  = blockIdx.x >> 5;
  const int q0 = (blockIdx.x & 31) * 64 + wave * 16;

  // Q^T A-fragments for the full 16x256 tile, kept in registers
  const _Float16* qrow = Qh + ((size_t)b * Ll + q0 + (lane & 15)) * DK;
  v16h qa[8];
#pragma unroll
  for (int dc = 0; dc < 8; dc++) qa[dc] = load_frag_a(qrow + dc * 32, lane);

  v8f zero = {};
  v8f acc[16];
#pragma unroll
  for (int t = 0; t < 16; t++) acc[t] = zero;
  float mrun[8], lrun[8];
#pragma unroll
  for (int r = 0; r < 8; r++) { mrun[r] = -3.0e38f; lrun[r] = 0.0f; }

  _Float16* pl = pLds[wave];
  const float sc = 0.0625f;  // 1/sqrt(DK)

  auto issue_tiles = [&](int buf, int kt) {
    // K tile: rows kt..kt+31 of Kh[b][l][d] -> Kt[buf] (fully contiguous)
    tdm_load_2d((unsigned int)(size_t)(&Kt[buf][0]),
                (unsigned long long)(size_t)(Kh + ((size_t)b * Ll + kt) * DK),
                /*tensor_d0=*/DK, /*tensor_d1=*/Ll,
                /*tile_d0=*/DK, /*tile_d1=*/32, /*stride0=*/DK);
    // V tile: cols kt..kt+31 of Vh[b][d][l] -> Vt[buf] (256 rows of 32)
    tdm_load_2d((unsigned int)(size_t)(&Vt[buf][0]),
                (unsigned long long)(size_t)(Vh + (size_t)b * DK * Ll + kt),
                /*tensor_d0=*/Ll, /*tensor_d1=*/DK,
                /*tile_d0=*/32, /*tile_d1=*/DK, /*stride0=*/Ll);
  };

  if (wave == 0) issue_tiles(0, 0);
  int p = 0;

  for (int kt = 0; kt < Ll; kt += 32) {
    if (wave == 0) __builtin_amdgcn_s_wait_tensorcnt(0);
    __syncthreads();   // buffer p ready; all waves done reading buffer p^1
    if (wave == 0 && kt + 32 < Ll) issue_tiles(p ^ 1, kt + 32);

    // S = Q^T K for 32 keys (two 16x16 tiles), B-fragments from LDS
    v8f s0 = zero, s1 = zero;
    const _Float16* kb = &Kt[p][0];
    const _Float16* kcol0 = kb + (size_t)(lane & 15) * DK;
    const _Float16* kcol1 = kb + (size_t)(16 + (lane & 15)) * DK;
#pragma unroll
    for (int dc = 0; dc < 8; dc++) {
      v16h b0 = load_frag_b(kcol0 + dc * 32, lane);
      s0 = wmma16(qa[dc], b0, s0);
      v16h b1 = load_frag_b(kcol1 + dc * 32, lane);
      s1 = wmma16(qa[dc], b1, s1);
    }

    // online softmax; rows live in 16-lane halves of each accum VGPR
    float corr[8];
#pragma unroll
    for (int r = 0; r < 8; r++) {
      float v0 = s0[r] * sc, v1 = s1[r] * sc;
      float mx = fmaxf(v0, v1);
      mx = fmaxf(mx, __shfl_xor(mx, 1, 16));
      mx = fmaxf(mx, __shfl_xor(mx, 2, 16));
      mx = fmaxf(mx, __shfl_xor(mx, 4, 16));
      mx = fmaxf(mx, __shfl_xor(mx, 8, 16));
      float mnew = fmaxf(mrun[r], mx);
      float p0 = __expf(v0 - mnew);
      float p1 = __expf(v1 - mnew);
      float s = p0 + p1;
      s += __shfl_xor(s, 1, 16);
      s += __shfl_xor(s, 2, 16);
      s += __shfl_xor(s, 4, 16);
      s += __shfl_xor(s, 8, 16);
      float c = __expf(mrun[r] - mnew);
      lrun[r] = lrun[r] * c + s;
      mrun[r] = mnew;
      corr[r] = c;
      const int q = r + ((lane >= 16) ? 8 : 0);
      pl[q * 32 + (lane & 15)]      = (_Float16)p0;
      pl[q * 32 + (lane & 15) + 16] = (_Float16)p1;
    }
    // re-layout P (16x32) through LDS into an A-fragment; DS ops from the
    // same wave are in-order, the wait covers the VGPR hazard.
    asm volatile("s_wait_dscnt 0" ::: "memory");
    v16h pa = load_frag_a(pl + (lane & 15) * 32, lane);
    asm volatile("s_wait_dscnt 0" ::: "memory");

    // acc = acc*corr + P x V^T  (16 d-tiles, V^T fragments from LDS)
#pragma unroll
    for (int t = 0; t < 16; t++) {
      const _Float16* vcol = &Vt[p][(size_t)(t * 16 + (lane & 15)) * 32];
      v16h vb = load_frag_b(vcol, lane);
      v8f a = acc[t];
#pragma unroll
      for (int r = 0; r < 8; r++) a[r] *= corr[r];
      acc[t] = wmma16(pa, vb, a);
    }
    p ^= 1;
  }

  // epilogue: divide by softmax denom, store H[b][d][q] (contiguous over r)
  const int hi = (lane >= 16) ? 8 : 0;
#pragma unroll
  for (int t = 0; t < 16; t++) {
    const int d = t * 16 + (lane & 15);
    float* dst = H + ((size_t)b * DK + d) * Ll + q0 + hi;
#pragma unroll
    for (int r = 0; r < 8; r++) dst[r] = acc[t][r] / lrun[r];
  }
}

// ---- launch --------------------------------------------------------------

extern "C" void kernel_launch(void* const* d_in, const int* in_sizes, int n_in,
                              void* d_out, int out_size, void* d_ws, size_t ws_size,
                              hipStream_t stream) {
  const float* x  = (const float*)d_in[0];
  const float* Wq = (const float*)d_in[1];
  const float* bq = (const float*)d_in[2];
  const float* Wk = (const float*)d_in[3];
  const float* bk = (const float*)d_in[4];
  const float* Wv = (const float*)d_in[5];
  const float* bv = (const float*)d_in[6];
  float* H = (float*)d_out;

  char* ws = (char*)d_ws;
  size_t off = 0;
  _Float16* Wh  = (_Float16*)(ws + off); off += (size_t)3 * DK * KDIM * 2;   // 1.18 MB
  _Float16* xim = (_Float16*)(ws + off); off += (size_t)Bn * Ll * KDIM * 2;  // 25.2 MB
  _Float16* Qh  = (_Float16*)(ws + off); off += (size_t)Bn * Ll * DK * 2;    // 8.4 MB
  _Float16* Kh  = (_Float16*)(ws + off); off += (size_t)Bn * Ll * DK * 2;    // 8.4 MB
  _Float16* Vh  = (_Float16*)(ws + off); off += (size_t)Bn * Ll * DK * 2;    // 8.4 MB
  (void)ws_size; (void)in_sizes; (void)n_in; (void)out_size;

  pack_w_kernel<<<768, 256, 0, stream>>>(Wq, Wk, Wv, Wh);
  pack_x_kernel<<<8192, 256, 0, stream>>>(x, xim);

  dim3 gridC(Ll / 64, DK / 64, 3 * Bn);  // (32, 4, 24)
  conv_gemm_kernel<<<gridC, 256, 0, stream>>>(Wh, xim, bq, bk, bv, Qh, Kh, Vh);

  attn_kernel<<<Bn * (Ll / 64), 128, 0, stream>>>(Qh, Kh, Vh, H);
}
